// GraphMamba_53652731462299
// MI455X (gfx1250) — compile-verified, matrix operand order checked by reading, not run on hardware
//
#include <hip/hip_runtime.h>
#include <hip/hip_bf16.h>
#include <cstdint>
#include <cstddef>

// ---------------------------------------------------------------------------
// Problem constants (from the reference)
// ---------------------------------------------------------------------------
#define TT   8
#define NN   2048
#define HH   256
#define PPOS 256
#define EE   512
#define SS   16
#define RR   16

typedef __attribute__((ext_vector_type(16))) __bf16 v16bf;
typedef __attribute__((ext_vector_type(8)))  float  v8f;
typedef __attribute__((ext_vector_type(8)))  __bf16 bf8;
typedef __attribute__((ext_vector_type(4)))  float  f32x4;
typedef __attribute__((ext_vector_type(4)))  int    i32x4;

// ---------------------------------------------------------------------------
// CDNA5 async global->LDS copy (ASYNCcnt-tracked) with sync fallback.
// Signature probe-confirmed: param0 = int4 addrspace(1)*, param1 = LDS ptr.
// ---------------------------------------------------------------------------
#if defined(__has_builtin)
#  if __has_builtin(__builtin_amdgcn_global_load_async_to_lds_b128) && \
      __has_builtin(__builtin_amdgcn_s_wait_asynccnt)
#    define HAVE_ASYNC 1
#  endif
#endif
#ifndef HAVE_ASYNC
#  define HAVE_ASYNC 0
#endif

__device__ __forceinline__ void cp16_async(const __bf16* g, __bf16* l)
{
#if HAVE_ASYNC
    __builtin_amdgcn_global_load_async_to_lds_b128(
        (__attribute__((address_space(1))) i32x4*)g,
        (__attribute__((address_space(3))) i32x4*)l, 0, 0);
#else
    *(bf8*)l = *(const bf8*)g;
#endif
}

__device__ __forceinline__ void wait_async_zero()
{
#if HAVE_ASYNC
    __builtin_amdgcn_s_wait_asynccnt(0);
#endif
}

// ---------------------------------------------------------------------------
// Shared epilogue helper
// ---------------------------------------------------------------------------
__device__ __forceinline__ float epi_act(float v, int act, float alpha, float beta)
{
    if      (act == 1) v = fmaxf(v, 0.f);
    else if (act == 2) v = 1.f / (1.f + __expf(-(v * alpha + beta)));
    else if (act == 3) v = v / (1.f + __expf(-v));
    else if (act == 4) v = (v > 20.f) ? v : log1pf(__expf(v));
    return v;
}

// ---------------------------------------------------------------------------
// Big GEMM: LDS double-buffered, block tile 64x128, 8 waves (2x4),
// each wave 2x2 WMMA tiles (32x32).  C[M,Nc] = act(A@B + bias) (+res).
// B supplied transposed (Bt[Nc,K] row-major).
// AMODE: 0 = A bf16;  1 = A = bf16 G * f32 ADJ (fused);  2 = A f32 converted.
// ---------------------------------------------------------------------------
template<int AMODE>
__global__ __launch_bounds__(256) void k_gemm_big(
    const __bf16* __restrict__ A, const float* __restrict__ A32,
    const float* __restrict__ ADJ,
    const __bf16* __restrict__ Bt,
    int M, int Nc, int K, int lda, int ldbt,
    const float* __restrict__ bias, int act, float alpha,
    const float* __restrict__ betaPtr,
    const float* __restrict__ res, int ldres,
    float* __restrict__ outF, __bf16* __restrict__ outB, int ldc,
    __bf16* __restrict__ outBT, int ldct)
{
    constexpr int LS = 40;                         // padded LDS row stride (bf16)
    __shared__ __bf16 sA[2][64 * LS];
    __shared__ __bf16 sB[2][128 * LS];

    const int tid  = threadIdx.x;
    const int lane = tid & 31;
    const int widx = tid >> 5;
    const int wm   = widx & 1;                     // wave row    (0..1)
    const int wn   = widx >> 1;                    // wave column (0..3)
    const int nb   = Nc >> 7;
    const int bm   = blockIdx.x / nb;
    const int bn   = blockIdx.x - bm * nb;
    const int M0   = bm << 6;
    const int N0   = bn << 7;
    const int half = lane >> 4;                    // K-chunk selector (ISA layout)
    const int r    = lane & 15;

    const int arow = tid >> 2;                     // A staging: 64 rows x 4 chunks
    const int acol = tid & 3;

    v8f acc[2][2];
#pragma unroll
    for (int ti = 0; ti < 2; ++ti)
#pragma unroll
        for (int tj = 0; tj < 2; ++tj)
#pragma unroll
            for (int i = 0; i < 8; ++i) acc[ti][tj][i] = 0.f;

    auto stageA = [&](int buf, int k0) {
        __bf16* l = &sA[buf][arow * LS + acol * 8];
        const size_t off = (size_t)(M0 + arow) * (size_t)lda + k0 + acol * 8;
        if constexpr (AMODE == 0) {
            cp16_async(A + off, l);
        } else if constexpr (AMODE == 1) {
            bf8   g0 = *(const bf8*)(A + off);
            f32x4 a0 = *(const f32x4*)(ADJ + off);
            f32x4 a1 = *(const f32x4*)(ADJ + off + 4);
            bf8 p;
#pragma unroll
            for (int i = 0; i < 4; ++i) {
                p[i]     = (__bf16)((float)g0[i]     * a0[i]);
                p[4 + i] = (__bf16)((float)g0[4 + i] * a1[i]);
            }
            *(bf8*)l = p;
        } else {
            f32x4 a0 = *(const f32x4*)(A32 + off);
            f32x4 a1 = *(const f32x4*)(A32 + off + 4);
            bf8 p;
#pragma unroll
            for (int i = 0; i < 4; ++i) {
                p[i]     = (__bf16)a0[i];
                p[4 + i] = (__bf16)a1[i];
            }
            *(bf8*)l = p;
        }
    };
    auto stageB = [&](int buf, int k0) {
#pragma unroll
        for (int i = 0; i < 2; ++i) {
            const int idx = tid + 256 * i;         // 128 rows x 4 chunks
            const int row = idx >> 2, cc = idx & 3;
            __bf16* l = &sB[buf][row * LS + cc * 8];
            const size_t off = (size_t)(N0 + row) * (size_t)ldbt + k0 + cc * 8;
            cp16_async(Bt + off, l);
        }
    };

    stageA(0, 0);
    stageB(0, 0);
    wait_async_zero();
    __syncthreads();

    const int nsteps = K >> 5;
    for (int s = 0; s < nsteps; ++s) {
        const int cur = s & 1;
        if (s + 1 < nsteps) {                      // prefetch next strip into LDS
            stageA(cur ^ 1, (s + 1) << 5);
            stageB(cur ^ 1, (s + 1) << 5);
        }
        v16bf af[2], bfv[2];
#pragma unroll
        for (int ti = 0; ti < 2; ++ti) {
            const __bf16* p = &sA[cur][(wm * 32 + ti * 16 + r) * LS + half * 8];
            bf8 c0 = *(const bf8*)p;
            bf8 c1 = *(const bf8*)(p + 16);
#pragma unroll
            for (int i = 0; i < 8; ++i) { af[ti][i] = c0[i]; af[ti][8 + i] = c1[i]; }
        }
#pragma unroll
        for (int tj = 0; tj < 2; ++tj) {
            const __bf16* p = &sB[cur][(wn * 32 + tj * 16 + r) * LS + half * 8];
            bf8 c0 = *(const bf8*)p;
            bf8 c1 = *(const bf8*)(p + 16);
#pragma unroll
            for (int i = 0; i < 8; ++i) { bfv[tj][i] = c0[i]; bfv[tj][8 + i] = c1[i]; }
        }
#pragma unroll
        for (int ti = 0; ti < 2; ++ti)
#pragma unroll
            for (int tj = 0; tj < 2; ++tj)
                acc[ti][tj] = __builtin_amdgcn_wmma_f32_16x16x32_bf16(
                    false, af[ti], false, bfv[tj], (short)0, acc[ti][tj],
                    false, false);
        wait_async_zero();
        __syncthreads();
    }

    // Epilogue: VGPR rr -> lanes 0-15: (M=rr, N=lane); lanes 16-31: (M=rr+8).
    const float beta = betaPtr ? *betaPtr : 0.f;
#pragma unroll
    for (int ti = 0; ti < 2; ++ti) {
#pragma unroll
        for (int tj = 0; tj < 2; ++tj) {
            const int n  = N0 + wn * 32 + tj * 16 + r;
            const float bv = bias ? bias[n] : 0.f;
#pragma unroll
            for (int rr = 0; rr < 8; ++rr) {
                const int m = M0 + wm * 32 + ti * 16 + rr + half * 8;
                float v = epi_act(acc[ti][tj][rr] + bv, act, alpha, beta);
                if (res)   v += res[(size_t)m * (size_t)ldres + n];
                if (outF)  outF[(size_t)m * (size_t)ldc + n] = v;
                if (outB)  outB[(size_t)m * (size_t)ldc + n] = (__bf16)v;
                if (outBT) outBT[(size_t)n * (size_t)ldct + m] = (__bf16)v;
            }
        }
    }
}

// ---------------------------------------------------------------------------
// Small GEMM (for Nc==16 rank-16 projections): one 16x16 tile per wave.
// ---------------------------------------------------------------------------
template<int AMODE>
__global__ __launch_bounds__(256) void k_gemm(
    const __bf16* __restrict__ A, const float* __restrict__ A32,
    const float* __restrict__ ADJ,
    const __bf16* __restrict__ Bt,
    int M, int Nc, int K, int lda, int ldbt,
    const float* __restrict__ bias, int act, float alpha,
    const float* __restrict__ betaPtr,
    const float* __restrict__ res, int ldres,
    float* __restrict__ outF, __bf16* __restrict__ outB, int ldc,
    __bf16* __restrict__ outBT, int ldct)
{
    const int lane    = threadIdx.x & 31;
    const int widx    = threadIdx.x >> 5;
    const int tiles_n = Nc >> 4;
    const int tiles_m = M >> 4;
    const int tile    = blockIdx.x * 8 + widx;
    if (tile >= tiles_m * tiles_n) return;
    const int tm   = tile / tiles_n;
    const int tn   = tile - tm * tiles_n;
    const int m0   = tm << 4;
    const int n0   = tn << 4;
    const int half = lane >> 4;
    const int r    = lane & 15;

    v8f acc;
#pragma unroll
    for (int i = 0; i < 8; ++i) acc[i] = 0.f;

    const __bf16* bp = Bt + (size_t)(n0 + r) * (size_t)ldbt + half * 8;

    for (int k0 = 0; k0 < K; k0 += 32) {
        v16bf af, bfv;
        {
            bf8 c0 = *(const bf8*)(bp + k0);
            bf8 c1 = *(const bf8*)(bp + k0 + 16);
#pragma unroll
            for (int i = 0; i < 8; ++i) { bfv[i] = c0[i]; bfv[8 + i] = c1[i]; }
        }
        if constexpr (AMODE == 0) {
            const __bf16* ap = A + (size_t)(m0 + r) * (size_t)lda + k0 + half * 8;
            bf8 c0 = *(const bf8*)ap;
            bf8 c1 = *(const bf8*)(ap + 16);
#pragma unroll
            for (int i = 0; i < 8; ++i) { af[i] = c0[i]; af[8 + i] = c1[i]; }
        } else {
            const size_t off = (size_t)(m0 + r) * (size_t)lda + k0 + half * 8;
            f32x4 a0 = *(const f32x4*)(A32 + off);
            f32x4 a1 = *(const f32x4*)(A32 + off + 4);
            f32x4 a2 = *(const f32x4*)(A32 + off + 16);
            f32x4 a3 = *(const f32x4*)(A32 + off + 20);
#pragma unroll
            for (int i = 0; i < 4; ++i) {
                af[i]      = (__bf16)a0[i];
                af[4 + i]  = (__bf16)a1[i];
                af[8 + i]  = (__bf16)a2[i];
                af[12 + i] = (__bf16)a3[i];
            }
        }
        acc = __builtin_amdgcn_wmma_f32_16x16x32_bf16(
            false, af, false, bfv, (short)0, acc, false, false);
    }

    const float beta = betaPtr ? *betaPtr : 0.f;
    const int   n    = n0 + r;
    const float bv   = bias ? bias[n] : 0.f;
#pragma unroll
    for (int rr = 0; rr < 8; ++rr) {
        const int m = m0 + rr + half * 8;
        float v = epi_act(acc[rr] + bv, act, alpha, beta);
        if (res)   v += res[(size_t)m * (size_t)ldres + n];
        if (outF)  outF[(size_t)m * (size_t)ldc + n] = v;
        if (outB)  outB[(size_t)m * (size_t)ldc + n] = (__bf16)v;
        if (outBT) outBT[(size_t)n * (size_t)ldct + m] = (__bf16)v;
    }
}

// ---------------------------------------------------------------------------
// LayerNorm over rows of 256: one wave (32 lanes x 8 elems) per row.
// ---------------------------------------------------------------------------
__global__ __launch_bounds__(256) void k_ln(
    const float* __restrict__ in, int ldin, int rows,
    const float* __restrict__ g, const float* __restrict__ b,
    float* __restrict__ outF, int ldF, __bf16* __restrict__ outB, int ldB)
{
    const int row  = blockIdx.x * 8 + (threadIdx.x >> 5);
    const int lane = threadIdx.x & 31;
    if (row >= rows) return;
    const float* rp = in + (size_t)row * (size_t)ldin;
    float v[8], s = 0.f;
#pragma unroll
    for (int i = 0; i < 8; ++i) { v[i] = rp[lane + 32 * i]; s += v[i]; }
#pragma unroll
    for (int m = 16; m >= 1; m >>= 1) s += __shfl_xor(s, m, 32);
    const float mean = s * (1.f / 256.f);
    float var = 0.f;
#pragma unroll
    for (int i = 0; i < 8; ++i) { float d = v[i] - mean; var += d * d; }
#pragma unroll
    for (int m = 16; m >= 1; m >>= 1) var += __shfl_xor(var, m, 32);
    const float rstd = rsqrtf(var * (1.f / 256.f) + 1e-5f);
#pragma unroll
    for (int i = 0; i < 8; ++i) {
        const int idx = lane + 32 * i;
        const float o = (v[i] - mean) * rstd * g[idx] + b[idx];
        if (outF) outF[(size_t)row * (size_t)ldF + idx] = o;
        if (outB) outB[(size_t)row * (size_t)ldB + idx] = (__bf16)o;
    }
}

// ---------------------------------------------------------------------------
// Selective-scan: one block per node, each lane owns 2 channels x 16 states.
// ---------------------------------------------------------------------------
__global__ __launch_bounds__(256) void k_scan(
    const __bf16* __restrict__ u, const float* __restrict__ delta,
    const float* __restrict__ Bm, const float* __restrict__ Cm,
    const float* __restrict__ Aexp, float* __restrict__ y)
{
    const int n   = blockIdx.x;
    const int tid = threadIdx.x;
    const int e0  = tid * 2;
    float a0[SS], a1[SS], h0[SS], h1[SS];
#pragma unroll
    for (int s = 0; s < SS; ++s) {
        a0[s] = Aexp[e0 * SS + s];
        a1[s] = Aexp[(e0 + 1) * SS + s];
        h0[s] = 0.f; h1[s] = 0.f;
    }
    __shared__ float sB[SS], sC[SS];
    for (int t = 0; t < TT; ++t) {
        const size_t row = (size_t)t * NN + n;
        __syncthreads();
        if (tid < SS)           sB[tid]      = Bm[row * SS + tid];
        else if (tid < 2 * SS)  sC[tid - SS] = Cm[row * SS + (tid - SS)];
        __syncthreads();
        const float d0 = delta[row * EE + e0];
        const float d1 = delta[row * EE + e0 + 1];
        const float u0 = (float)u[row * EE + e0];
        const float u1 = (float)u[row * EE + e0 + 1];
        const float du0 = d0 * u0, du1 = d1 * u1;
        float y0 = 0.f, y1 = 0.f;
#pragma unroll
        for (int s = 0; s < SS; ++s) {
            h0[s] = __expf(d0 * a0[s]) * h0[s] + du0 * sB[s];
            h1[s] = __expf(d1 * a1[s]) * h1[s] + du1 * sB[s];
            y0 += h0[s] * sC[s];
            y1 += h1[s] * sC[s];
        }
        y[row * EE + e0]     = y0;
        y[row * EE + e0 + 1] = y1;
    }
}

// ---------------------------------------------------------------------------
// Small elementwise kernels
// ---------------------------------------------------------------------------
__global__ void k_pe(__bf16* __restrict__ pe)
{
    const int idx = blockIdx.x * blockDim.x + threadIdx.x;
    if (idx >= NN * PPOS) return;
    const int n = idx >> 8, d = idx & 255, j = d >> 1;
    const float div = __expf((float)(2 * j) * (-9.210340371976184f / 256.f));
    const float ang = (float)n * div;
    pe[idx] = (__bf16)((d & 1) ? __cosf(ang) : __sinf(ang));
}

__global__ void k_transpose(const float* __restrict__ in, __bf16* __restrict__ out,
                            int R, int C, int ldo)
{
    const int idx = blockIdx.x * blockDim.x + threadIdx.x;
    if (idx >= C * ldo) return;
    const int c = idx / ldo, r = idx - c * ldo;
    out[idx] = (r < R) ? (__bf16)in[(size_t)r * C + c] : (__bf16)0.f;
}

__global__ void k_copy_pe(const __bf16* __restrict__ pe, __bf16* __restrict__ cat1)
{
    const int idx = blockIdx.x * blockDim.x + threadIdx.x;
    if (idx >= NN * PPOS) return;
    const int m = idx >> 8, c = idx & 255;
    cat1[(size_t)m * 512 + c] = pe[idx];
}

__global__ void k_zero_bf16(__bf16* __restrict__ p, int n)
{
    const int idx = blockIdx.x * blockDim.x + threadIdx.x;
    if (idx < n) p[idx] = (__bf16)0.f;
}

__global__ void k_negexp(const float* __restrict__ in, float* __restrict__ out, int n)
{
    const int idx = blockIdx.x * blockDim.x + threadIdx.x;
    if (idx < n) out[idx] = -__expf(in[idx]);
}

__global__ void k_y2(const float* __restrict__ y, const __bf16* __restrict__ u,
                     const __bf16* __restrict__ sg, const float* __restrict__ D,
                     __bf16* __restrict__ y2, int total)
{
    const int idx = blockIdx.x * blockDim.x + threadIdx.x;
    if (idx >= total) return;
    const int e = idx & (EE - 1);
    y2[idx] = (__bf16)((y[idx] + (float)u[idx] * D[e]) * (float)sg[idx]);
}

// ---------------------------------------------------------------------------
// Host-side GEMM dispatch
// ---------------------------------------------------------------------------
static inline void gemm(hipStream_t s, int amode,
                        const __bf16* A, const float* A32, const float* ADJ,
                        const __bf16* Bt, int M, int Nc, int K, int lda, int ldbt,
                        const float* bias, int act, float alpha, const float* betaPtr,
                        const float* res, int ldres,
                        float* outF, __bf16* outB, int ldc,
                        __bf16* outBT, int ldct)
{
    dim3 block(256);
    if ((M % 64 == 0) && (Nc % 128 == 0) && (K % 32 == 0)) {
        dim3 grid((unsigned)((M / 64) * (Nc / 128)));
        if (amode == 0)
            k_gemm_big<0><<<grid, block, 0, s>>>(A, A32, ADJ, Bt, M, Nc, K, lda, ldbt,
                bias, act, alpha, betaPtr, res, ldres, outF, outB, ldc, outBT, ldct);
        else if (amode == 1)
            k_gemm_big<1><<<grid, block, 0, s>>>(A, A32, ADJ, Bt, M, Nc, K, lda, ldbt,
                bias, act, alpha, betaPtr, res, ldres, outF, outB, ldc, outBT, ldct);
        else
            k_gemm_big<2><<<grid, block, 0, s>>>(A, A32, ADJ, Bt, M, Nc, K, lda, ldbt,
                bias, act, alpha, betaPtr, res, ldres, outF, outB, ldc, outBT, ldct);
    } else {
        const int tiles = (M / 16) * (Nc / 16);
        dim3 grid((unsigned)((tiles + 7) / 8));
        if (amode == 2)
            k_gemm<2><<<grid, block, 0, s>>>(A, A32, ADJ, Bt, M, Nc, K, lda, ldbt,
                bias, act, alpha, betaPtr, res, ldres, outF, outB, ldc, outBT, ldct);
        else
            k_gemm<0><<<grid, block, 0, s>>>(A, A32, ADJ, Bt, M, Nc, K, lda, ldbt,
                bias, act, alpha, betaPtr, res, ldres, outF, outB, ldc, outBT, ldct);
    }
}

extern "C" void kernel_launch(void* const* d_in, const int* in_sizes, int n_in,
                              void* d_out, int out_size, void* d_ws, size_t ws_size,
                              hipStream_t stream)
{
    (void)in_sizes; (void)n_in; (void)out_size; (void)ws_size;
    const float* adj      = (const float*)d_in[0];
    const float* g1_msg_w = (const float*)d_in[1];
    const float* g1_msg_b = (const float*)d_in[2];
    const float* g1_q_w   = (const float*)d_in[3];
    const float* g1_k_w   = (const float*)d_in[4];
    const float* g1_gateb = (const float*)d_in[5];
    const float* g1_upd_w = (const float*)d_in[6];
    const float* g1_upd_b = (const float*)d_in[7];
    const float* g1_ln_g  = (const float*)d_in[8];
    const float* g1_ln_b  = (const float*)d_in[9];
    const float* g2_msg_w = (const float*)d_in[10];
    const float* g2_msg_b = (const float*)d_in[11];
    const float* g2_upd_w = (const float*)d_in[12];
    const float* g2_upd_b = (const float*)d_in[13];
    const float* g2_ln_g  = (const float*)d_in[14];
    const float* g2_ln_b  = (const float*)d_in[15];
    const float* m_ln_g   = (const float*)d_in[16];
    const float* m_ln_b   = (const float*)d_in[17];
    const float* m_in_w   = (const float*)d_in[18];
    const float* m_in_b   = (const float*)d_in[19];
    const float* m_dl_w   = (const float*)d_in[20];
    const float* m_dl_b   = (const float*)d_in[21];
    const float* m_dt_w   = (const float*)d_in[22];
    const float* m_dt_b   = (const float*)d_in[23];
    const float* m_B_w    = (const float*)d_in[24];
    const float* m_B_b    = (const float*)d_in[25];
    const float* m_C_w    = (const float*)d_in[26];
    const float* m_C_b    = (const float*)d_in[27];
    const float* m_Alog   = (const float*)d_in[28];
    const float* m_D      = (const float*)d_in[29];
    const float* m_out_w  = (const float*)d_in[30];
    const float* m_out_b  = (const float*)d_in[31];
    float* out = (float*)d_out;                      // [T,N,H], row = t*N+n

    char* wp = (char*)d_ws;
    auto carve = [&](size_t bytes) -> void* {
        void* r = (void*)wp;
        wp += (bytes + 255) & ~(size_t)255;
        return r;
    };
    const int MTN = TT * NN;
    __bf16* peB     = (__bf16*)carve((size_t)NN * PPOS * 2);
    __bf16* w1msgT  = (__bf16*)carve((size_t)HH * PPOS * 2);
    __bf16* w1qT    = (__bf16*)carve((size_t)HH * HH * 2);
    __bf16* w1kT    = (__bf16*)carve((size_t)HH * HH * 2);
    __bf16* w1updT  = (__bf16*)carve((size_t)HH * 512 * 2);
    __bf16* w2msgT  = (__bf16*)carve((size_t)HH * HH * 2);
    __bf16* w2updT  = (__bf16*)carve((size_t)HH * 512 * 2);
    __bf16* minT    = (__bf16*)carve((size_t)1024 * HH * 2);
    __bf16* wdlT    = (__bf16*)carve((size_t)RR * EE * 2);
    __bf16* wdtT    = (__bf16*)carve((size_t)EE * 32 * 2);
    __bf16* wBT     = (__bf16*)carve((size_t)SS * EE * 2);
    __bf16* wCT     = (__bf16*)carve((size_t)SS * EE * 2);
    __bf16* woutT   = (__bf16*)carve((size_t)HH * EE * 2);
    __bf16* msgs1B  = (__bf16*)carve((size_t)NN * HH * 2);
    __bf16* msgs1T  = (__bf16*)carve((size_t)HH * NN * 2);
    __bf16* qB      = (__bf16*)carve((size_t)NN * HH * 2);
    __bf16* kB      = (__bf16*)carve((size_t)NN * HH * 2);
    __bf16* gatesB  = (__bf16*)carve((size_t)NN * NN * 2);
    __bf16* cat1    = (__bf16*)carve((size_t)NN * 512 * 2);
    __bf16* cat2    = (__bf16*)carve((size_t)NN * 512 * 2);
    __bf16* msgs2T  = (__bf16*)carve((size_t)HH * NN * 2);
    float*  h1      = (float*) carve((size_t)NN * HH * 4);
    float*  h2      = (float*) carve((size_t)NN * HH * 4);
    float*  x2      = (float*) carve((size_t)MTN * HH * 4);
    __bf16* xnB     = (__bf16*)carve((size_t)MTN * HH * 2);
    __bf16* uB      = (__bf16*)carve((size_t)MTN * EE * 2);
    __bf16* sgB     = (__bf16*)carve((size_t)MTN * EE * 2);
    __bf16* dlPad   = (__bf16*)carve((size_t)MTN * 32 * 2);
    float*  deltaF  = (float*) carve((size_t)MTN * EE * 4);
    float*  BmF     = (float*) carve((size_t)MTN * SS * 4);
    float*  CmF     = (float*) carve((size_t)MTN * SS * 4);
    float*  AexpF   = (float*) carve((size_t)EE * SS * 4);
    float*  yF      = (float*) carve((size_t)MTN * EE * 4);
    __bf16* y2B     = (__bf16*)carve((size_t)MTN * EE * 2);

    dim3 b256(256);
    auto g1d = [](int n) { return dim3((unsigned)((n + 255) / 256)); };
    auto tr = [&](const float* in, __bf16* o, int R, int C, int ldo) {
        k_transpose<<<g1d(C * ldo), b256, 0, stream>>>(in, o, R, C, ldo);
    };

    // ---- one-time prep ----
    k_pe<<<g1d(NN * PPOS), b256, 0, stream>>>(peB);
    tr(g1_msg_w, w1msgT, PPOS, HH, PPOS);
    tr(g1_q_w,   w1qT,   HH, HH, HH);
    tr(g1_k_w,   w1kT,   HH, HH, HH);
    tr(g1_upd_w, w1updT, 512, HH, 512);
    tr(g2_msg_w, w2msgT, HH, HH, HH);
    tr(g2_upd_w, w2updT, 512, HH, 512);
    tr(m_in_w,   minT,   HH, 1024, HH);
    tr(m_dl_w,   wdlT,   EE, RR, EE);
    tr(m_dt_w,   wdtT,   RR, EE, 32);
    tr(m_B_w,    wBT,    EE, SS, EE);
    tr(m_C_w,    wCT,    EE, SS, EE);
    tr(m_out_w,  woutT,  EE, HH, EE);
    k_negexp<<<g1d(EE * SS), b256, 0, stream>>>(m_Alog, AexpF, EE * SS);
    k_zero_bf16<<<g1d(MTN * 32), b256, 0, stream>>>(dlPad, MTN * 32);
    k_copy_pe<<<g1d(NN * PPOS), b256, 0, stream>>>(peB, cat1);

    // ---- timestep-invariant GNN-1 front half ----
    gemm(stream, 0, peB, nullptr, nullptr, w1msgT, NN, HH, PPOS, PPOS, PPOS,
         g1_msg_b, 0, 0.f, nullptr, nullptr, 0, nullptr, msgs1B, HH, msgs1T, NN);
    gemm(stream, 0, msgs1B, nullptr, nullptr, w1qT, NN, HH, HH, HH, HH,
         nullptr, 0, 0.f, nullptr, nullptr, 0, nullptr, qB, HH, nullptr, 0);
    gemm(stream, 0, msgs1B, nullptr, nullptr, w1kT, NN, HH, HH, HH, HH,
         nullptr, 0, 0.f, nullptr, nullptr, 0, nullptr, kB, HH, nullptr, 0);
    gemm(stream, 0, qB, nullptr, nullptr, kB, NN, NN, HH, HH, HH,
         nullptr, 2, 0.0625f, g1_gateb, nullptr, 0, nullptr, gatesB, NN, nullptr, 0);

    // ---- per-timestep GNN layers ----
    for (int t = 0; t < TT; ++t) {
        const float* adjT = adj + (size_t)t * NN * NN;
        gemm(stream, 1, gatesB, nullptr, adjT, msgs1T, NN, HH, NN, NN, NN,
             nullptr, 0, 0.f, nullptr, nullptr, 0, nullptr, cat1 + PPOS, 512, nullptr, 0);
        gemm(stream, 0, cat1, nullptr, nullptr, w1updT, NN, HH, 512, 512, 512,
             g1_upd_b, 1, 0.f, nullptr, nullptr, 0, h1, nullptr, HH, nullptr, 0);
        k_ln<<<dim3(NN / 8), b256, 0, stream>>>(h1, HH, NN, g1_ln_g, g1_ln_b,
                                                nullptr, 0, cat2, 512);
        gemm(stream, 0, cat2, nullptr, nullptr, w2msgT, NN, HH, HH, 512, HH,
             g2_msg_b, 0, 0.f, nullptr, nullptr, 0, nullptr, nullptr, 0, msgs2T, NN);
        gemm(stream, 2, nullptr, adjT, nullptr, msgs2T, NN, HH, NN, NN, NN,
             nullptr, 0, 0.f, nullptr, nullptr, 0, nullptr, cat2 + PPOS, 512, nullptr, 0);
        gemm(stream, 0, cat2, nullptr, nullptr, w2updT, NN, HH, 512, 512, 512,
             g2_upd_b, 1, 0.f, nullptr, nullptr, 0, h2, nullptr, HH, nullptr, 0);
        k_ln<<<dim3(NN / 8), b256, 0, stream>>>(h2, HH, NN, g2_ln_g, g2_ln_b,
                                                x2 + (size_t)t * NN * HH, HH, nullptr, 0);
    }

    // ---- Mamba over rows (t*N+n) ----
    k_ln<<<dim3(MTN / 8), b256, 0, stream>>>(x2, HH, MTN, m_ln_g, m_ln_b,
                                             nullptr, 0, xnB, HH);
    gemm(stream, 0, xnB, nullptr, nullptr, minT, MTN, EE, HH, HH, HH,
         m_in_b, 3, 0.f, nullptr, nullptr, 0, nullptr, uB, EE, nullptr, 0);
    gemm(stream, 0, xnB, nullptr, nullptr, minT + (size_t)EE * HH, MTN, EE, HH, HH, HH,
         m_in_b + EE, 3, 0.f, nullptr, nullptr, 0, nullptr, sgB, EE, nullptr, 0);
    gemm(stream, 0, uB, nullptr, nullptr, wdlT, MTN, RR, EE, EE, EE,
         m_dl_b, 0, 0.f, nullptr, nullptr, 0, nullptr, dlPad, 32, nullptr, 0);
    gemm(stream, 0, dlPad, nullptr, nullptr, wdtT, MTN, EE, 32, 32, 32,
         m_dt_b, 4, 0.f, nullptr, nullptr, 0, deltaF, nullptr, EE, nullptr, 0);
    gemm(stream, 0, uB, nullptr, nullptr, wBT, MTN, SS, EE, EE, EE,
         m_B_b, 0, 0.f, nullptr, nullptr, 0, BmF, nullptr, SS, nullptr, 0);
    gemm(stream, 0, uB, nullptr, nullptr, wCT, MTN, SS, EE, EE, EE,
         m_C_b, 0, 0.f, nullptr, nullptr, 0, CmF, nullptr, SS, nullptr, 0);
    k_scan<<<dim3(NN), b256, 0, stream>>>(uB, deltaF, BmF, CmF, AexpF, yF);
    k_y2<<<g1d(MTN * EE), b256, 0, stream>>>(yF, uB, sgB, m_D, y2B, MTN * EE);
    gemm(stream, 0, y2B, nullptr, nullptr, woutT, MTN, HH, EE, EE, EE,
         m_out_b, 0, 0.f, nullptr, x2, HH, out, nullptr, HH, nullptr, 0);
}
